// WhereValuePredictor_35845797053209
// MI455X (gfx1250) — matrix operand order repaired and checked by reading, not compiled, not run on hardware
//
#include <hip/hip_runtime.h>
#include <math.h>

typedef float  v8f   __attribute__((ext_vector_type(8)));
typedef __bf16 v16bf __attribute__((ext_vector_type(16)));
typedef __bf16 v8bf  __attribute__((ext_vector_type(8)));
typedef __bf16 v4bf  __attribute__((ext_vector_type(4)));

__device__ __forceinline__ __bf16 f2bf(float f) {
    union { float f; unsigned u; } v; v.f = f;
    unsigned r = (v.u + 0x7fffu + ((v.u >> 16) & 1u)) >> 16;
    union { unsigned short s; __bf16 b; } o; o.s = (unsigned short)r;
    return o.b;
}
__device__ __forceinline__ float sigf(float x) { return 1.0f / (1.0f + __expf(-x)); }
__device__ __forceinline__ v8f zero8() {
    v8f z;
    #pragma unroll
    for (int i = 0; i < 8; ++i) z[i] = 0.0f;
    return z;
}

// ---- fp32 -> bf16 bulk convert (n multiple of 1024), grid = n/1024 --------
__global__ __launch_bounds__(256) void cvt_kernel(const float* __restrict__ src,
                                                  __bf16* __restrict__ dst)
{
    const size_t i = ((size_t)blockIdx.x * 256 + threadIdx.x) * 4;
    float4 v = *(const float4*)(src + i);
    v4bf o;
    o[0] = f2bf(v.x); o[1] = f2bf(v.y); o[2] = f2bf(v.z); o[3] = f2bf(v.w);
    *(v4bf*)(dst + i) = o;
}

// ---------------------------------------------------------------------------
// bf16 WMMA GEMM: C[m,n] = sum_k A[m,k]*Bw[n,k] + bias[n]
// A bf16 (M x K, lda), Bw bf16 row-major (N x K, ldb), C fp32 (ldc).
// Block: 256 thr = 8 waves; wave computes 16x64 (4 n-tiles); block 16 x 512.
// Grid: (M/16, N/512). K multiple of 32.
// ---------------------------------------------------------------------------
__global__ __launch_bounds__(256) void gemm_bf16_kernel(
    const __bf16* __restrict__ A, int lda,
    const __bf16* __restrict__ Bw, int ldb,
    const float* __restrict__ bias,
    float* __restrict__ C, int ldc, int K)
{
    const int lane = threadIdx.x & 31, wave = threadIdx.x >> 5;
    const int lm = lane & 15, lh = lane >> 4;
    const int m0 = blockIdx.x * 16;
    const int n0 = blockIdx.y * 512 + wave * 64;

    const __bf16* arow = A + (size_t)(m0 + lm) * lda;
    v8f acc[4] = {zero8(), zero8(), zero8(), zero8()};

    for (int K0 = 0; K0 < K; K0 += 32) {
        v8bf a0 = *(const v8bf*)(arow + K0 + lh * 16);
        v8bf a1 = *(const v8bf*)(arow + K0 + lh * 16 + 8);
        v16bf a = __builtin_shufflevector(a0, a1, 0, 1, 2, 3, 4, 5, 6, 7,
                                          8, 9, 10, 11, 12, 13, 14, 15);
        #pragma unroll
        for (int nt = 0; nt < 4; ++nt) {
            const __bf16* brow = Bw + (size_t)(n0 + nt * 16 + lm) * ldb + K0 + lh * 16;
            v16bf b = *(const v16bf*)brow;
            acc[nt] = __builtin_amdgcn_wmma_f32_16x16x32_bf16(false, a, false, b,
                                                              (short)0, acc[nt], false, false);
        }
    }
    #pragma unroll
    for (int nt = 0; nt < 4; ++nt) {
        const int n = n0 + nt * 16 + lm;
        const float bs = bias ? bias[n] : 0.0f;
        #pragma unroll
        for (int r = 0; r < 8; ++r)
            C[(size_t)(m0 + r + 8 * lh) * ldc + n] = acc[nt][r] + bs;
    }
}

// ---------------------------------------------------------------------------
// Persistent LSTM recurrence for one (layer,dir).
// Block = 16 batch rows, 256 thr (8 waves). h double-buffered in LDS (bf16),
// c in registers. Per step: gates(16x2048) = h(16x512) @ Whh^T via WMMA.
// pre fp32 (Btot*T, 2048); whh bf16 (2048,512); out fp32 (Btot,T,1024) halves.
// ---------------------------------------------------------------------------
__global__ __launch_bounds__(256) void lstm_kernel(
    const float* __restrict__ pre,
    const __bf16* __restrict__ whh,
    float* __restrict__ out, int T, int dir)
{
    __shared__ __bf16 hsh[2][16][512];
    const int lane = threadIdx.x & 31, wave = threadIdx.x >> 5;
    const int lm = lane & 15, lh = lane >> 4;
    const int b0 = blockIdx.x * 16;
    const int dofs = dir ? 512 : 0;

    for (int i = threadIdx.x; i < 16 * 512; i += 256)
        hsh[0][i >> 9][i & 511] = f2bf(0.0f);
    float c_reg[32];
    #pragma unroll
    for (int i = 0; i < 32; ++i) c_reg[i] = 0.0f;
    __syncthreads();

    for (int s = 0; s < T; ++s) {
        const int t = dir ? (T - 1 - s) : s;
        const int cur = s & 1, nxt = cur ^ 1;
        for (int jt = 0; jt < 4; ++jt) {
            const int j0 = wave * 64 + jt * 16;
            v8f acc[4] = {zero8(), zero8(), zero8(), zero8()};
            for (int kk = 0; kk < 16; ++kk) {
                const int K0 = kk * 32;
                const int ks = K0 + lh * 16;
                v8bf a0 = *(const v8bf*)&hsh[cur][lm][ks];
                v8bf a1 = *(const v8bf*)&hsh[cur][lm][ks + 8];
                v16bf a = __builtin_shufflevector(a0, a1, 0, 1, 2, 3, 4, 5, 6, 7,
                                                  8, 9, 10, 11, 12, 13, 14, 15);
                #pragma unroll
                for (int g = 0; g < 4; ++g) {
                    const __bf16* brow = whh + (size_t)(g * 512 + j0 + lm) * 512 + K0 + lh * 16;
                    v16bf b = *(const v16bf*)brow;
                    acc[g] = __builtin_amdgcn_wmma_f32_16x16x32_bf16(false, a, false, b,
                                                                     (short)0, acc[g], false, false);
                }
            }
            const int jc = j0 + lm;
            #pragma unroll
            for (int r = 0; r < 8; ++r) {
                const int m = r + 8 * lh;
                const int row = b0 + m;
                const float* prow = pre + ((size_t)row * T + t) * 2048;
                float iv = acc[0][r] + prow[jc];
                float fv = acc[1][r] + prow[512 + jc];
                float gv = acc[2][r] + prow[1024 + jc];
                float ov = acc[3][r] + prow[1536 + jc];
                float c = sigf(fv) * c_reg[jt * 8 + r] + sigf(iv) * tanhf(gv);
                c_reg[jt * 8 + r] = c;
                float h = sigf(ov) * tanhf(c);
                hsh[nxt][m][jc] = f2bf(h);
                out[((size_t)row * T + t) * 1024 + dofs + jc] = h;
            }
        }
        __syncthreads();
    }
}

// ---- q_mask: 1 wave per (b,t) row -----------------------------------------
__global__ __launch_bounds__(256) void qmask_kernel(const float* __restrict__ x,
                                                    float* __restrict__ mask, int rows)
{
    const int lane = threadIdx.x & 31, wv = threadIdx.x >> 5;
    const int row = blockIdx.x * 8 + wv;
    if (row >= rows) return;
    const float* p = x + (size_t)row * 1024;
    float s = 0.0f;
    for (int h = lane; h < 1024; h += 32) s += fabsf(p[h]);
    for (int o = 16; o > 0; o >>= 1) s += __shfl_down(s, o);
    if (lane == 0) mask[row] = (s != 0.0f) ? 1.0f : 0.0f;
}

// ---- column softmax pooling: 1 block per packed column p ------------------
__global__ __launch_bounds__(256) void colpool_kernel(
    const float* __restrict__ cemb, const float* __restrict__ cenc,
    const float* __restrict__ w, const float* __restrict__ bptr,
    float* __restrict__ pooled)
{
    __shared__ float logit[8], attn[8];
    const int p = blockIdx.x;
    const int lane = threadIdx.x & 31, wv = threadIdx.x >> 5;
    const float* xrow = cemb + ((size_t)p * 8 + wv) * 1024;
    const float* erow = cenc + ((size_t)p * 8 + wv) * 1024;
    float asum = 0.0f, dot = 0.0f;
    for (int h = lane; h < 1024; h += 32) { asum += fabsf(xrow[h]); dot += erow[h] * w[h]; }
    for (int o = 16; o > 0; o >>= 1) { asum += __shfl_down(asum, o); dot += __shfl_down(dot, o); }
    if (lane == 0) logit[wv] = (asum != 0.0f) ? (dot + bptr[0]) : -INFINITY;
    __syncthreads();
    if (threadIdx.x < 8) {
        float mx = -INFINITY;
        for (int t = 0; t < 8; ++t) mx = fmaxf(mx, logit[t]);
        float den = 0.0f;
        for (int t = 0; t < 8; ++t) den += (logit[t] == -INFINITY) ? 0.0f : __expf(logit[t] - mx);
        float l = logit[threadIdx.x];
        attn[threadIdx.x] = (l == -INFINITY || den == 0.0f) ? 0.0f : __expf(l - mx) / den;
    }
    __syncthreads();
    for (int h = threadIdx.x; h < 1024; h += 256) {
        float s = 0.0f;
        for (int t = 0; t < 8; ++t) s += attn[t] * cenc[((size_t)p * 8 + t) * 1024 + h];
        pooled[(size_t)p * 1024 + h] = s;
    }
}

// ---- gather observed columns ----------------------------------------------
__global__ __launch_bounds__(256) void gather_kernel(
    const float* __restrict__ pooled, const int* __restrict__ gold_cols,
    const int* __restrict__ gold_num, float* __restrict__ observed)
{
    const int bk = blockIdx.x, b = bk >> 2, k = bk & 3;
    int gn = gold_num[b]; if (gn > 4) gn = 4;
    const float cm = (k < gn) ? 1.0f : 0.0f;
    const int col = gold_cols[bk];
    const float* src = pooled + ((size_t)b * 20 + col) * 1024;
    for (int h = threadIdx.x; h < 1024; h += 256)
        observed[(size_t)bk * 1024 + h] = cm * src[h];
}

// ---- attention over question: 1 block per (b,k) ---------------------------
__global__ __launch_bounds__(256) void attn_kernel(
    const float* __restrict__ attkey, const float* __restrict__ qenc,
    const float* __restrict__ qmask, float* __restrict__ qhid)
{
    __shared__ float sc[128], attn[128];
    const int bk = blockIdx.x, b = bk >> 2;
    const int lane = threadIdx.x & 31, wv = threadIdx.x >> 5;
    const float* key = attkey + (size_t)bk * 1024;
    for (int q = wv; q < 128; q += 8) {
        const float* e = qenc + ((size_t)b * 128 + q) * 1024;
        float dot = 0.0f;
        for (int h = lane; h < 1024; h += 32) dot += key[h] * e[h];
        for (int o = 16; o > 0; o >>= 1) dot += __shfl_down(dot, o);
        if (lane == 0) sc[q] = (qmask[b * 128 + q] > 0.0f) ? dot : -INFINITY;
    }
    __syncthreads();
    float mx = -INFINITY;
    for (int q = 0; q < 128; ++q) mx = fmaxf(mx, sc[q]);
    float den = 0.0f;
    for (int q = 0; q < 128; ++q) den += (sc[q] == -INFINITY) ? 0.0f : __expf(sc[q] - mx);
    for (int q = threadIdx.x; q < 128; q += 256)
        attn[q] = (sc[q] == -INFINITY || mx == -INFINITY || den == 0.0f)
                      ? 0.0f : __expf(sc[q] - mx) / den;
    __syncthreads();
    for (int h = threadIdx.x; h < 1024; h += 256) {
        float s = 0.0f;
        for (int q = 0; q < 128; ++q) s += attn[q] * qenc[((size_t)b * 128 + q) * 1024 + h];
        qhid[(size_t)bk * 1024 + h] = s;
    }
}

// ---- fused tanh(U+V+b1) -> out2 dot -> mask -> pos0/pos1 ------------------
__global__ __launch_bounds__(256) void final_kernel(
    const float* __restrict__ U, const float* __restrict__ V,
    const float* __restrict__ w2, const float* __restrict__ b2,
    const float* __restrict__ qmask, const int* __restrict__ gold_num,
    float* __restrict__ out)
{
    const int bk = blockIdx.x, b = bk >> 2, k = bk & 3;
    const int lane = threadIdx.x & 31, wv = threadIdx.x >> 5;
    int gn = gold_num[b]; if (gn > 4) gn = 4;
    const bool cm = (k < gn);
    const float* vrow = V + (size_t)bk * 1024;
    for (int q = wv; q < 128; q += 8) {
        const float* urow = U + ((size_t)b * 128 + q) * 1024;
        float s0 = 0.0f, s1 = 0.0f;
        for (int h = lane; h < 1024; h += 32) {
            float y = tanhf(urow[h] + vrow[h]);
            s0 += y * w2[h];
            s1 += y * w2[1024 + h];
        }
        for (int o = 16; o > 0; o >>= 1) { s0 += __shfl_down(s0, o); s1 += __shfl_down(s1, o); }
        if (lane == 0) {
            const bool m = cm && (qmask[b * 128 + q] > 0.0f);
            out[(size_t)bk * 128 + q]         = m ? (s0 + b2[0]) : -INFINITY;
            out[16384 + (size_t)bk * 128 + q] = m ? (s1 + b2[1]) : -INFINITY;
        }
    }
}

// ---------------------------------------------------------------------------
extern "C" void kernel_launch(void* const* d_in, const int* in_sizes, int n_in,
                              void* d_out, int out_size, void* d_ws, size_t ws_size,
                              hipStream_t stream)
{
    (void)in_sizes; (void)n_in; (void)out_size; (void)ws_size;
    const float* q_emb      = (const float*)d_in[0];
    const float* c_emb      = (const float*)d_in[1];
    const int*   gold_num   = (const int*)d_in[2];
    const int*   gold_cols  = (const int*)d_in[3];
    const float* q_Wih      = (const float*)d_in[6];
    const float* q_Whh      = (const float*)d_in[7];
    const float* q_bias     = (const float*)d_in[8];
    const float* c_Wih      = (const float*)d_in[9];
    const float* c_Whh      = (const float*)d_in[10];
    const float* c_bias     = (const float*)d_in[11];
    const float* col_attn_w = (const float*)d_in[12];
    const float* col_attn_b = (const float*)d_in[13];
    const float* ob_attn_W  = (const float*)d_in[14];
    const float* ob_attn_b  = (const float*)d_in[15];
    const float* ob_col_W   = (const float*)d_in[16];
    const float* ob_col_b   = (const float*)d_in[17];
    const float* q_proj_W   = (const float*)d_in[18];
    const float* q_proj_b   = (const float*)d_in[19];
    const float* out1_W     = (const float*)d_in[20];
    const float* out1_b     = (const float*)d_in[21];
    const float* out2_W     = (const float*)d_in[22];
    const float* out2_b     = (const float*)d_in[23];
    float* out = (float*)d_out;

    char* wsb = (char*)d_ws;
    size_t off = 0;
    auto alloc = [&](size_t bytes) {
        off = (off + 255) & ~(size_t)255;
        void* p = wsb + off;
        off += bytes;
        return p;
    };
    // fp32 buffers
    float* pre      = (float*)alloc((size_t)5120 * 2048 * 4);  // reused as U later
    float* qenc0    = (float*)alloc((size_t)4096 * 1024 * 4);
    float* qenc1    = (float*)alloc((size_t)4096 * 1024 * 4);
    float* cenc0    = (float*)alloc((size_t)5120 * 1024 * 4);
    float* cenc1    = (float*)alloc((size_t)5120 * 1024 * 4);
    float* pooled   = (float*)alloc((size_t)640 * 1024 * 4);
    float* observed = (float*)alloc((size_t)128 * 1024 * 4);
    float* attkey   = (float*)alloc((size_t)128 * 1024 * 4);
    float* qhid     = (float*)alloc((size_t)128 * 1024 * 4);
    float* vec      = (float*)alloc((size_t)128 * 2048 * 4);
    float* Vb       = (float*)alloc((size_t)128 * 1024 * 4);
    float* qmask    = (float*)alloc((size_t)4096 * 4);
    float* U        = pre;  // alias: pre is dead after the LSTM stages
    // bf16 buffers
    __bf16* wq_ih   = (__bf16*)alloc((size_t)4 * 2048 * 1024 * 2);
    __bf16* wq_hh   = (__bf16*)alloc((size_t)4 * 2048 * 512 * 2);
    __bf16* wc_ih   = (__bf16*)alloc((size_t)4 * 2048 * 1024 * 2);
    __bf16* wc_hh   = (__bf16*)alloc((size_t)4 * 2048 * 512 * 2);
    __bf16* w_oba   = (__bf16*)alloc((size_t)1024 * 1024 * 2);
    __bf16* w_obc   = (__bf16*)alloc((size_t)1024 * 1024 * 2);
    __bf16* w_qpr   = (__bf16*)alloc((size_t)1024 * 1024 * 2);
    __bf16* w_o1    = (__bf16*)alloc((size_t)1024 * 3072 * 2);
    __bf16* Abf     = (__bf16*)alloc((size_t)5120 * 1024 * 2);  // shared A operand

    const dim3 blk(256);
    auto cvt = [&](const float* s, __bf16* d, size_t n) {
        cvt_kernel<<<dim3((unsigned)(n / 1024)), blk, 0, stream>>>(s, d);
    };

    // one-time weight conversion to bf16
    cvt(q_Wih, wq_ih, (size_t)4 * 2048 * 1024);
    cvt(q_Whh, wq_hh, (size_t)4 * 2048 * 512);
    cvt(c_Wih, wc_ih, (size_t)4 * 2048 * 1024);
    cvt(c_Whh, wc_hh, (size_t)4 * 2048 * 512);
    cvt(ob_attn_W, w_oba, (size_t)1024 * 1024);
    cvt(ob_col_W, w_obc, (size_t)1024 * 1024);
    cvt(q_proj_W, w_qpr, (size_t)1024 * 1024);
    cvt(out1_W, w_o1, (size_t)1024 * 3072);

    qmask_kernel<<<512, blk, 0, stream>>>(q_emb, qmask, 4096);

    // question BiLSTM (B=32, T=128)
    for (int l = 0; l < 2; ++l) {
        float* o = (l == 0) ? qenc0 : qenc1;
        cvt((l == 0) ? q_emb : qenc0, Abf, (size_t)4096 * 1024);
        for (int d = 0; d < 2; ++d) {
            const int wsel = l * 2 + d;
            gemm_bf16_kernel<<<dim3(4096 / 16, 4), blk, 0, stream>>>(
                Abf, 1024, wq_ih + (size_t)wsel * 2048 * 1024, 1024,
                q_bias + (size_t)wsel * 2048, pre, 2048, 1024);
            lstm_kernel<<<dim3(32 / 16), blk, 0, stream>>>(
                pre, wq_hh + (size_t)wsel * 2048 * 512, o, 128, d);
        }
    }
    // column BiLSTM (B=640, T=8)
    for (int l = 0; l < 2; ++l) {
        float* o = (l == 0) ? cenc0 : cenc1;
        cvt((l == 0) ? c_emb : cenc0, Abf, (size_t)5120 * 1024);
        for (int d = 0; d < 2; ++d) {
            const int wsel = l * 2 + d;
            gemm_bf16_kernel<<<dim3(5120 / 16, 4), blk, 0, stream>>>(
                Abf, 1024, wc_ih + (size_t)wsel * 2048 * 1024, 1024,
                c_bias + (size_t)wsel * 2048, pre, 2048, 1024);
            lstm_kernel<<<dim3(640 / 16), blk, 0, stream>>>(
                pre, wc_hh + (size_t)wsel * 2048 * 512, o, 8, d);
        }
    }

    colpool_kernel<<<640, blk, 0, stream>>>(c_emb, cenc1, col_attn_w, col_attn_b, pooled);
    gather_kernel<<<128, blk, 0, stream>>>(pooled, gold_cols, gold_num, observed);

    // observed -> att_key and ob_col projections
    cvt(observed, Abf, (size_t)128 * 1024);
    gemm_bf16_kernel<<<dim3(8, 2), blk, 0, stream>>>(Abf, 1024, w_oba, 1024,
                                                     ob_attn_b, attkey, 1024, 1024);
    gemm_bf16_kernel<<<dim3(8, 2), blk, 0, stream>>>(Abf, 1024, w_obc, 1024,
                                                     ob_col_b, vec, 2048, 1024);
    attn_kernel<<<128, blk, 0, stream>>>(attkey, qenc1, qmask, qhid);

    cvt(qhid, Abf, (size_t)128 * 1024);
    gemm_bf16_kernel<<<dim3(8, 2), blk, 0, stream>>>(Abf, 1024, w_qpr, 1024,
                                                     q_proj_b, vec + 1024, 2048, 1024);

    // factored out1: U = q_enc @ W1^T ; V = vec @ W2^T + b1
    cvt(qenc1, Abf, (size_t)4096 * 1024);
    gemm_bf16_kernel<<<dim3(4096 / 16, 2), blk, 0, stream>>>(Abf, 1024, w_o1, 3072,
                                                             nullptr, U, 1024, 1024);
    cvt(vec, Abf, (size_t)128 * 2048);
    gemm_bf16_kernel<<<dim3(8, 2), blk, 0, stream>>>(Abf, 2048, w_o1 + 1024, 3072,
                                                     out1_b, Vb, 1024, 2048);

    final_kernel<<<128, blk, 0, stream>>>(U, Vb, out2_W, out2_b, qmask, gold_num, out);
}
